// PointNetRot9d_40931038331358
// MI455X (gfx1250) — matrix operand order repaired and checked by reading, not compiled, not run on hardware
//
#include <hip/hip_runtime.h>
#include <hip/hip_bf16.h>
#include <math.h>

// ---------------------------------------------------------------------------
// CDNA5 WMMA (wave32): v_wmma_f32_16x16x32_bf16  D(16x16 f32) = A(16x32) x B(32x16) + C
// A frag: lane l -> row m0+(l&15), K run  (l>>4)*16, pairs packed per dword
// B frag: lane l -> col n0+(l&15), K run  (l>>4)*16, pairs packed per dword
// D frag: VGPR r -> row r+8*(l>>4), col (l&15)
// ---------------------------------------------------------------------------
typedef __attribute__((ext_vector_type(16))) __bf16 v16bf;
typedef __attribute__((ext_vector_type(8)))  float  v8f;

#define PPTS 65536   // B*N = 32*2048
#define NPT  2048
#define EPSBN 1e-5f

__device__ __forceinline__ unsigned short f2bf(float f) {        // RNE f32->bf16
    unsigned int u = __float_as_uint(f);
    u += 0x7FFFu + ((u >> 16) & 1u);
    return (unsigned short)(u >> 16);
}
__device__ __forceinline__ float bf2f(unsigned short h) {
    return __uint_as_float((unsigned int)h << 16);
}

// A fragment from row-major bf16 weights: two contiguous 16B loads per lane
__device__ __forceinline__ v16bf load_a_tile(const unsigned short* wrow_k) {
    union { v16bf v; uint4 q[2]; } A;
    A.q[0] = *reinterpret_cast<const uint4*>(wrow_k);
    A.q[1] = *reinterpret_cast<const uint4*>(wrow_k + 8);
    return A.v;
}

// B fragment from pre-swizzled activations: two contiguous 16B loads per lane
__device__ __forceinline__ v16bf load_b_frag(const uint4* bb) {
    union { v16bf v; uint4 q[2]; } B;
    B.q[0] = bb[0];
    B.q[1] = bb[1];
    return B.v;
}

// Swizzled activation layout (B-fragment order), K = #channels (64 or 128):
//   dword index = ntile*(K*8) + t*256 + lane*8 + i
//   holds bf16 pair (k = t*32 + (lane>>4)*16 + 2i [,+1]) of point p = ntile*16 + (lane&15)

// ---------------------------------------------------------------------------
// Utility kernels
// ---------------------------------------------------------------------------
__global__ void convert_bf16_kernel(const float* __restrict__ src,
                                    unsigned short* __restrict__ dst, int n) {
    int i = blockIdx.x * blockDim.x + threadIdx.x;
    if (i < n) dst[i] = f2bf(src[i]);
}

__global__ void zero_kernel(float* __restrict__ p, int n) {
    int i = blockIdx.x * blockDim.x + threadIdx.x;
    if (i < n) p[i] = 0.0f;
}

// Layer 1: 3 -> 64 scalar (K=3), writes bf16 pre-act [64, P]. (b1 cancels in BN)
__global__ void conv1_kernel(const float* __restrict__ x, const float* __restrict__ w1,
                             unsigned short* __restrict__ out) {
    int idx = blockIdx.x * blockDim.x + threadIdx.x;   // 64 * 65536
    int c = idx >> 16;
    int p = idx & 65535;
    int b = p >> 11;
    int n = p & 2047;
    const float* xp = x + (size_t)b * 3 * NPT + n;
    float v = w1[c * 3 + 0] * xp[0] + w1[c * 3 + 1] * xp[NPT] + w1[c * 3 + 2] * xp[2 * NPT];
    out[(size_t)c * PPTS + p] = f2bf(v);
}

// per-channel mean / rsqrt(var+eps) over P for bf16 pre [C, P]; one block per channel
__global__ void rowstats_bf16_kernel(const unsigned short* __restrict__ pre,
                                     float* __restrict__ mean, float* __restrict__ rstd, int P) {
    __shared__ float ss[256], sq[256];
    int c = blockIdx.x;
    const unsigned short* row = pre + (size_t)c * P;
    float s = 0.f, q = 0.f;
    for (int i = threadIdx.x; i < P; i += 256) { float v = bf2f(row[i]); s += v; q += v * v; }
    ss[threadIdx.x] = s; sq[threadIdx.x] = q;
    __syncthreads();
    for (int o = 128; o > 0; o >>= 1) {
        if (threadIdx.x < o) { ss[threadIdx.x] += ss[threadIdx.x + o]; sq[threadIdx.x] += sq[threadIdx.x + o]; }
        __syncthreads();
    }
    if (threadIdx.x == 0) {
        float m = ss[0] / (float)P;
        float var = sq[0] / (float)P - m * m;
        mean[c] = m;
        rstd[c] = rsqrtf(var + EPSBN);
    }
}

// BN affine + ReLU, writing the SWIZZLED B-fragment layout. One thread per dword
// of the destination (stores perfectly coalesced); kb = log2(K*8).
__global__ void bn_relu_swz_kernel(const unsigned short* __restrict__ pre,  // [C,P] bf16
                                   const float* __restrict__ g, const float* __restrict__ be,
                                   const float* __restrict__ mean, const float* __restrict__ rstd,
                                   unsigned int* __restrict__ dst, int P, int kb, int total) {
    int idx = blockIdx.x * blockDim.x + threadIdx.x;
    if (idx >= total) return;
    int ntile = idx >> kb;
    int rem   = idx & ((1 << kb) - 1);
    int t     = rem >> 8;
    int rem2  = rem & 255;
    int lane  = rem2 >> 3;
    int i     = rem2 & 7;
    int k = t * 32 + (lane >> 4) * 16 + 2 * i;
    int p = (ntile << 4) + (lane & 15);
    float sc0 = g[k] * rstd[k],         sh0 = be[k] - sc0 * mean[k];
    float sc1 = g[k + 1] * rstd[k + 1], sh1 = be[k + 1] - sc1 * mean[k + 1];
    float v0 = fmaxf(bf2f(pre[(size_t)k * P + p]) * sc0 + sh0, 0.0f);
    float v1 = fmaxf(bf2f(pre[(size_t)(k + 1) * P + p]) * sc1 + sh1, 0.0f);
    dst[idx] = (unsigned int)f2bf(v0) | ((unsigned int)f2bf(v1) << 16);
}

// ---------------------------------------------------------------------------
// Layer 2 GEMM: pre2[128,P] (bf16) = w2[128,64] * act1(swizzled). One wave per
// 16x64 output strip (A fragment reused across 4 n-tiles). Grid tiles exactly.
// ---------------------------------------------------------------------------
__global__ __launch_bounds__(256) void wmma_gemm2_kernel(
    const unsigned short* __restrict__ Wbf,   // [128,64] bf16 row-major
    const unsigned int* __restrict__ actS,    // swizzled, K=64 -> 512 dwords/tile
    unsigned short* __restrict__ out, int P) {
    int wid  = (blockIdx.x * blockDim.x + threadIdx.x) >> 5;  // 0..8191
    int lane = threadIdx.x & 31;
    int mT = wid >> 10;             // 8 row tiles
    int ng = wid & 1023;            // 1024 groups of 4 n-tiles
    int m0 = mT << 4;
    int lh = lane >> 4, lm = lane & 15;
    int arow = m0 + lm, akoff = lh * 16;
    v16bf A0 = load_a_tile(Wbf + (size_t)arow * 64 + 0  + akoff);
    v16bf A1 = load_a_tile(Wbf + (size_t)arow * 64 + 32 + akoff);
#pragma unroll
    for (int j = 0; j < 4; ++j) {
        int ntile = ng * 4 + j;
        const uint4* bb = reinterpret_cast<const uint4*>(actS + (size_t)ntile * 512 + lane * 8);
        v8f acc = {};
        acc = __builtin_amdgcn_wmma_f32_16x16x32_bf16(false, A0, false, load_b_frag(bb),      (short)0, acc, false, false);
        acc = __builtin_amdgcn_wmma_f32_16x16x32_bf16(false, A1, false, load_b_frag(bb + 64), (short)0, acc, false, false);
        int col = (ntile << 4) + lm;
#pragma unroll
        for (int r = 0; r < 8; ++r)
            out[(size_t)(m0 + r + 8 * lh) * P + col] = f2bf(acc[r]);
    }
}

// ---------------------------------------------------------------------------
// Layer 3 pass 1: GEMM (1024 x 65536 x 128), per-channel sum/sumsq only.
// wave = (channel tile, batch); w3 A fragments hoisted out of the 128-tile loop.
// ---------------------------------------------------------------------------
__global__ __launch_bounds__(256) void wmma_l3_stats_kernel(
    const unsigned short* __restrict__ Wbf,   // [1024,128] bf16
    const unsigned int* __restrict__ actS,    // swizzled, K=128 -> 1024 dwords/tile
    float* __restrict__ sum, float* __restrict__ sumsq) {
    int wid  = (blockIdx.x * blockDim.x + threadIdx.x) >> 5;  // 0..2047
    int lane = threadIdx.x & 31;
    int ct = wid >> 5;      // 0..63
    int b  = wid & 31;      // 0..31
    int c0 = ct << 4;
    int lh = lane >> 4, lm = lane & 15;
    int arow = c0 + lm, akoff = lh * 16;
    v16bf A[4];
#pragma unroll
    for (int t = 0; t < 4; ++t)
        A[t] = load_a_tile(Wbf + (size_t)arow * 128 + t * 32 + akoff);
    float s[8], q[8];
#pragma unroll
    for (int r = 0; r < 8; ++r) { s[r] = 0.f; q[r] = 0.f; }
    for (int nt = 0; nt < 128; ++nt) {
        const uint4* bb = reinterpret_cast<const uint4*>(
            actS + (size_t)(b * 128 + nt) * 1024 + lane * 8);
        v8f acc = {};
#pragma unroll
        for (int t = 0; t < 4; ++t)
            acc = __builtin_amdgcn_wmma_f32_16x16x32_bf16(false, A[t], false,
                    load_b_frag(bb + t * 64), (short)0, acc, false, false);
#pragma unroll
        for (int r = 0; r < 8; ++r) { float v = acc[r]; s[r] += v; q[r] += v * v; }
    }
#pragma unroll
    for (int r = 0; r < 8; ++r) {
        float sv = s[r], qv = q[r];
        for (int off = 8; off > 0; off >>= 1) {
            sv += __shfl_xor(sv, off, 16);
            qv += __shfl_xor(qv, off, 16);
        }
        if (lm == 0) {
            int c = c0 + r + 8 * lh;
            atomicAdd(&sum[c], sv);
            atomicAdd(&sumsq[c], qv);
        }
    }
}

__global__ void finalize_stats_kernel(const float* __restrict__ sum, const float* __restrict__ sumsq,
                                      float* __restrict__ mean, float* __restrict__ rstd,
                                      float invcnt, int C) {
    int c = blockIdx.x * blockDim.x + threadIdx.x;
    if (c >= C) return;
    float m = sum[c] * invcnt;
    float var = sumsq[c] * invcnt - m * m;
    mean[c] = m;
    rstd[c] = rsqrtf(var + EPSBN);
}

// ---------------------------------------------------------------------------
// Layer 3 pass 2: recompute GEMM, BN affine folded to scale/shift, fused global
// max-pool over N=2048 in registers; width-16 shfl max; writes pooled[b,c].
// ---------------------------------------------------------------------------
__global__ __launch_bounds__(256) void wmma_l3_maxpool_kernel(
    const unsigned short* __restrict__ Wbf, const unsigned int* __restrict__ actS,
    const float* __restrict__ g, const float* __restrict__ be,
    const float* __restrict__ mean, const float* __restrict__ rstd,
    float* __restrict__ pooled) {
    int wid  = (blockIdx.x * blockDim.x + threadIdx.x) >> 5;
    int lane = threadIdx.x & 31;
    int ct = wid >> 5;
    int b  = wid & 31;
    int c0 = ct << 4;
    int lh = lane >> 4, lm = lane & 15;
    int arow = c0 + lm, akoff = lh * 16;
    v16bf A[4];
#pragma unroll
    for (int t = 0; t < 4; ++t)
        A[t] = load_a_tile(Wbf + (size_t)arow * 128 + t * 32 + akoff);
    float scale[8], shift[8], mx[8];
#pragma unroll
    for (int r = 0; r < 8; ++r) {
        int c = c0 + r + 8 * lh;
        float gs = g[c] * rstd[c];
        scale[r] = gs;
        shift[r] = be[c] - gs * mean[c];
        mx[r] = -INFINITY;
    }
    for (int nt = 0; nt < 128; ++nt) {
        const uint4* bb = reinterpret_cast<const uint4*>(
            actS + (size_t)(b * 128 + nt) * 1024 + lane * 8);
        v8f acc = {};
#pragma unroll
        for (int t = 0; t < 4; ++t)
            acc = __builtin_amdgcn_wmma_f32_16x16x32_bf16(false, A[t], false,
                    load_b_frag(bb + t * 64), (short)0, acc, false, false);
#pragma unroll
        for (int r = 0; r < 8; ++r)
            mx[r] = fmaxf(mx[r], acc[r] * scale[r] + shift[r]);
    }
#pragma unroll
    for (int r = 0; r < 8; ++r) {
        float v = mx[r];
        for (int off = 8; off > 0; off >>= 1)
            v = fmaxf(v, __shfl_xor(v, off, 16));
        if (lm == 0)
            pooled[b * 1024 + c0 + r + 8 * lh] = v;
    }
}

// ---------------------------------------------------------------------------
// FC head (tiny, scalar f32). FC biases cancel in BN-over-batch -> skipped.
// ---------------------------------------------------------------------------
__global__ void fc_kernel(const float* __restrict__ in, const float* __restrict__ w,
                          float* __restrict__ out, int K, int F, int total) {
    int i = blockIdx.x * blockDim.x + threadIdx.x;
    if (i >= total) return;
    int b = i / F, f = i - b * F;
    const float* ip = in + (size_t)b * K;
    const float* wp = w + (size_t)f * K;
    float s = 0.f;
    for (int k = 0; k < K; ++k) s += ip[k] * wp[k];
    out[i] = s;
}

__global__ void colstats_kernel(const float* __restrict__ pre,
                                float* __restrict__ mean, float* __restrict__ rstd, int F) {
    int f = blockIdx.x * blockDim.x + threadIdx.x;
    if (f >= F) return;
    float s = 0.f, q = 0.f;
    for (int b = 0; b < 32; ++b) { float v = pre[b * F + f]; s += v; q += v * v; }
    float m = s * (1.0f / 32.0f);
    float var = q * (1.0f / 32.0f) - m * m;
    mean[f] = m;
    rstd[f] = rsqrtf(var + EPSBN);
}

__global__ void bn_relu_fc_kernel(const float* __restrict__ pre,
                                  const float* __restrict__ g, const float* __restrict__ be,
                                  const float* __restrict__ mean, const float* __restrict__ rstd,
                                  float* __restrict__ out, int F, int total) {
    int i = blockIdx.x * blockDim.x + threadIdx.x;
    if (i >= total) return;
    int f = i % F;
    float sc = g[f] * rstd[f];
    out[i] = fmaxf((pre[i] - mean[f]) * sc + be[f], 0.0f);
}

__global__ void head_kernel(const float* __restrict__ in, const float* __restrict__ wp,
                            const float* __restrict__ bp, float* __restrict__ out) {
    int i = blockIdx.x * blockDim.x + threadIdx.x;    // 32*12
    if (i >= 32 * 12) return;
    int b = i / 12, o = i - b * 12;
    const float* ip = in + (size_t)b * 256;
    const float* w = wp + (size_t)o * 256;
    float s = bp[o];
    for (int k = 0; k < 256; ++k) s += ip[k] * w[k];
    out[i] = s;
}

// ---------------------------------------------------------------------------
// 3x3 symmetric orthogonalization (Jacobi eigen of M^T M), one thread/batch.
// ---------------------------------------------------------------------------
__device__ __forceinline__ float det3(const float m[3][3]) {
    return m[0][0] * (m[1][1] * m[2][2] - m[1][2] * m[2][1])
         - m[0][1] * (m[1][0] * m[2][2] - m[1][2] * m[2][0])
         + m[0][2] * (m[1][0] * m[2][1] - m[1][1] * m[2][0]);
}

__global__ void svd_assemble_kernel(const float* __restrict__ p, float* __restrict__ out) {
    int b = threadIdx.x;
    if (b >= 32) return;
    float M[3][3];
#pragma unroll
    for (int i = 0; i < 3; ++i)
#pragma unroll
        for (int j = 0; j < 3; ++j)
            M[i][j] = p[b * 12 + i * 3 + j];
    float tx = p[b * 12 + 9], ty = p[b * 12 + 10], tz = p[b * 12 + 11];

    float A[3][3];
#pragma unroll
    for (int i = 0; i < 3; ++i)
#pragma unroll
        for (int j = 0; j < 3; ++j)
            A[i][j] = M[0][i] * M[0][j] + M[1][i] * M[1][j] + M[2][i] * M[2][j];
    float V[3][3] = {{1, 0, 0}, {0, 1, 0}, {0, 0, 1}};

    const int PQ[3][2] = {{0, 1}, {0, 2}, {1, 2}};
    for (int sweep = 0; sweep < 12; ++sweep) {
#pragma unroll
        for (int r = 0; r < 3; ++r) {
            int pp = PQ[r][0], qq = PQ[r][1];
            float apq = A[pp][qq];
            if (fabsf(apq) > 1e-15f) {
                float theta = (A[qq][qq] - A[pp][pp]) / (2.0f * apq);
                float tt = ((theta >= 0.f) ? 1.f : -1.f) / (fabsf(theta) + sqrtf(theta * theta + 1.0f));
                float c = 1.0f / sqrtf(tt * tt + 1.0f);
                float s = tt * c;
#pragma unroll
                for (int k = 0; k < 3; ++k) {
                    float akp = A[k][pp], akq = A[k][qq];
                    A[k][pp] = c * akp - s * akq;
                    A[k][qq] = s * akp + c * akq;
                }
#pragma unroll
                for (int k = 0; k < 3; ++k) {
                    float apk = A[pp][k], aqk = A[qq][k];
                    A[pp][k] = c * apk - s * aqk;
                    A[qq][k] = s * apk + c * aqk;
                }
#pragma unroll
                for (int k = 0; k < 3; ++k) {
                    float vkp = V[k][pp], vkq = V[k][qq];
                    V[k][pp] = c * vkp - s * vkq;
                    V[k][qq] = s * vkp + c * vkq;
                }
            }
        }
    }
    float w[3] = {A[0][0], A[1][1], A[2][2]};
#pragma unroll
    for (int i = 0; i < 2; ++i)
#pragma unroll
        for (int j = 0; j < 2 - i; ++j)
            if (w[j] < w[j + 1]) {
                float tw = w[j]; w[j] = w[j + 1]; w[j + 1] = tw;
#pragma unroll
                for (int k = 0; k < 3; ++k) { float tv = V[k][j]; V[k][j] = V[k][j + 1]; V[k][j + 1] = tv; }
            }
    float U[3][3];
#pragma unroll
    for (int i = 0; i < 3; ++i) {
        float u0 = M[0][0] * V[0][i] + M[0][1] * V[1][i] + M[0][2] * V[2][i];
        float u1 = M[1][0] * V[0][i] + M[1][1] * V[1][i] + M[1][2] * V[2][i];
        float u2 = M[2][0] * V[0][i] + M[2][1] * V[1][i] + M[2][2] * V[2][i];
        float nn = sqrtf(u0 * u0 + u1 * u1 + u2 * u2);
        if (i == 2 && nn < 1e-12f) {
            u0 = U[1][0] * U[2][1] - U[2][0] * U[1][1];
            u1 = U[2][0] * U[0][1] - U[0][0] * U[2][1];
            u2 = U[0][0] * U[1][1] - U[1][0] * U[0][1];
            nn = sqrtf(u0 * u0 + u1 * u1 + u2 * u2);
        }
        float inv = 1.0f / fmaxf(nn, 1e-20f);
        U[0][i] = u0 * inv; U[1][i] = u1 * inv; U[2][i] = u2 * inv;
    }
    float d = (det3(U) * det3(V) >= 0.f) ? 1.0f : -1.0f;
    float R[3][3];
#pragma unroll
    for (int i = 0; i < 3; ++i)
#pragma unroll
        for (int j = 0; j < 3; ++j)
            R[i][j] = U[i][0] * V[j][0] + U[i][1] * V[j][1] + d * U[i][2] * V[j][2];

    float* o = out + b * 16;
#pragma unroll
    for (int i = 0; i < 3; ++i) {
        o[i * 4 + 0] = R[i][0]; o[i * 4 + 1] = R[i][1]; o[i * 4 + 2] = R[i][2];
    }
    o[3] = tx; o[7] = ty; o[11] = tz;
    o[12] = 0.f; o[13] = 0.f; o[14] = 0.f; o[15] = 1.f;
}

// ---------------------------------------------------------------------------
// Orchestration
// ---------------------------------------------------------------------------
extern "C" void kernel_launch(void* const* d_in, const int* in_sizes, int n_in,
                              void* d_out, int out_size, void* d_ws, size_t ws_size,
                              hipStream_t stream) {
    const float* x   = (const float*)d_in[0];
    const float* w1  = (const float*)d_in[1];
    const float* g1  = (const float*)d_in[3];
    const float* be1 = (const float*)d_in[4];
    const float* w2  = (const float*)d_in[5];
    const float* g2  = (const float*)d_in[7];
    const float* be2 = (const float*)d_in[8];
    const float* w3  = (const float*)d_in[9];
    const float* g3  = (const float*)d_in[11];
    const float* be3 = (const float*)d_in[12];
    const float* wf1 = (const float*)d_in[13];
    const float* gf1 = (const float*)d_in[15];
    const float* bef1= (const float*)d_in[16];
    const float* wf2 = (const float*)d_in[17];
    const float* gf2 = (const float*)d_in[19];
    const float* bef2= (const float*)d_in[20];
    const float* wp  = (const float*)d_in[21];
    const float* bp  = (const float*)d_in[22];
    float* out = (float*)d_out;

    char* ws = (char*)d_ws;
    size_t off = 0;
    auto take = [&](size_t bytes) { size_t r = off; off += (bytes + 255) & ~(size_t)255; return r; };

    unsigned short* w2bf  = (unsigned short*)(ws + take(128 * 64 * 2));
    unsigned short* w3bf  = (unsigned short*)(ws + take(1024 * 128 * 2));
    unsigned short* h1pre = (unsigned short*)(ws + take((size_t)64 * PPTS * 2));   // plain [64,P]
    unsigned int*   h1swz = (unsigned int*)  (ws + take((size_t)64 * PPTS * 2));   // swizzled
    unsigned short* h2pre = (unsigned short*)(ws + take((size_t)128 * PPTS * 2));  // plain [128,P]
    unsigned int*   h2swz = (unsigned int*)  (ws + take((size_t)128 * PPTS * 2));  // swizzled
    float* mean1 = (float*)(ws + take(64 * 4));
    float* rstd1 = (float*)(ws + take(64 * 4));
    float* mean2 = (float*)(ws + take(128 * 4));
    float* rstd2 = (float*)(ws + take(128 * 4));
    float* sum3  = (float*)(ws + take(1024 * 4));   // 4096B, 256-aligned ->
    float* sumsq3= (float*)(ws + take(1024 * 4));   // contiguous with sum3
    float* mean3 = (float*)(ws + take(1024 * 4));
    float* rstd3 = (float*)(ws + take(1024 * 4));
    float* pooled= (float*)(ws + take(32 * 1024 * 4));
    float* f1pre = (float*)(ws + take(32 * 512 * 4));
    float* f1act = (float*)(ws + take(32 * 512 * 4));
    float* meanf1= (float*)(ws + take(512 * 4));
    float* rstdf1= (float*)(ws + take(512 * 4));
    float* f2pre = (float*)(ws + take(32 * 256 * 4));
    float* f2act = (float*)(ws + take(32 * 256 * 4));
    float* meanf2= (float*)(ws + take(256 * 4));
    float* rstdf2= (float*)(ws + take(256 * 4));
    float* phead = (float*)(ws + take(32 * 12 * 4));
    (void)ws_size; (void)in_sizes; (void)n_in; (void)out_size;

    // weights -> bf16
    convert_bf16_kernel<<<(128 * 64 + 255) / 256, 256, 0, stream>>>(w2, w2bf, 128 * 64);
    convert_bf16_kernel<<<(1024 * 128 + 255) / 256, 256, 0, stream>>>(w3, w3bf, 1024 * 128);

    // Layer 1: scalar conv -> bf16 pre; stats; BN+ReLU -> swizzled (K=64, kb=9)
    conv1_kernel<<<(64 * PPTS) / 256, 256, 0, stream>>>(x, w1, h1pre);
    rowstats_bf16_kernel<<<64, 256, 0, stream>>>(h1pre, mean1, rstd1, PPTS);
    bn_relu_swz_kernel<<<(64 * PPTS / 2) / 256, 256, 0, stream>>>(
        h1pre, g1, be1, mean1, rstd1, h1swz, PPTS, 9, 64 * PPTS / 2);

    // Layer 2 WMMA GEMM (128 x 65536 x 64): 8192 waves -> 1024 blocks of 8
    wmma_gemm2_kernel<<<1024, 256, 0, stream>>>(w2bf, h1swz, h2pre, PPTS);
    rowstats_bf16_kernel<<<128, 256, 0, stream>>>(h2pre, mean2, rstd2, PPTS);
    bn_relu_swz_kernel<<<(128 * PPTS / 2) / 256, 256, 0, stream>>>(
        h2pre, g2, be2, mean2, rstd2, h2swz, PPTS, 10, 128 * PPTS / 2);

    // Layer 3: stats pass then fused BN+maxpool pass (no 268MB buffer)
    zero_kernel<<<(2048 + 255) / 256, 256, 0, stream>>>(sum3, 2048);  // sum3+sumsq3
    wmma_l3_stats_kernel<<<256, 256, 0, stream>>>(w3bf, h2swz, sum3, sumsq3);
    finalize_stats_kernel<<<4, 256, 0, stream>>>(sum3, sumsq3, mean3, rstd3,
                                                 1.0f / (float)PPTS, 1024);
    wmma_l3_maxpool_kernel<<<256, 256, 0, stream>>>(w3bf, h2swz, g3, be3, mean3, rstd3, pooled);

    // FC head (scalar f32)
    fc_kernel<<<(32 * 512 + 255) / 256, 256, 0, stream>>>(pooled, wf1, f1pre, 1024, 512, 32 * 512);
    colstats_kernel<<<2, 256, 0, stream>>>(f1pre, meanf1, rstdf1, 512);
    bn_relu_fc_kernel<<<(32 * 512 + 255) / 256, 256, 0, stream>>>(f1pre, gf1, bef1, meanf1, rstdf1,
                                                                  f1act, 512, 32 * 512);
    fc_kernel<<<(32 * 256 + 255) / 256, 256, 0, stream>>>(f1act, wf2, f2pre, 512, 256, 32 * 256);
    colstats_kernel<<<1, 256, 0, stream>>>(f2pre, meanf2, rstdf2, 256);
    bn_relu_fc_kernel<<<(32 * 256 + 255) / 256, 256, 0, stream>>>(f2pre, gf2, bef2, meanf2, rstdf2,
                                                                  f2act, 256, 32 * 256);
    head_kernel<<<2, 192, 0, stream>>>(f2act, wp, bp, phead);

    svd_assemble_kernel<<<1, 32, 0, stream>>>(phead, out);
}
